// LlamaForCausalLM_28613072126241
// MI455X (gfx1250) — compile-verified, matrix operand order checked by reading, not run on hardware
//
#include <hip/hip_runtime.h>

// ---- model constants ----
#define LL  4
#define DD  2048
#define HH  16
#define HDD 128
#define II  5632
#define VV  32000
#define BB  2
#define SS  1024
#define TT  (BB*SS)          // 2048 tokens

typedef __attribute__((ext_vector_type(16))) __bf16 bf16x16;
typedef __attribute__((ext_vector_type(8)))  float  f32x8;

union Frag {
  bf16x16 v;
  unsigned short u[16];
  uint4 q[2];
};

__device__ __forceinline__ unsigned short f2bf(float f) {
  __bf16 h = (__bf16)f;                      // backend native cvt
  return __builtin_bit_cast(unsigned short, h);
}
__device__ __forceinline__ float bf2f(unsigned short h) {
  return __uint_as_float(((unsigned int)h) << 16);
}

// ---------------------------------------------------------------------------
// fp32 -> bf16 bulk conversion (weight staging). n must be a multiple of 4.
// ---------------------------------------------------------------------------
__global__ __launch_bounds__(256) void cvt_f32_bf16(
    const float* __restrict__ src, unsigned short* __restrict__ dst, long n)
{
  const long i = ((long)blockIdx.x * 256 + threadIdx.x) * 4;
  if (i + 3 < n) {
    float4 f = *(const float4*)(src + i);
    uint2 o;
    o.x = (unsigned)f2bf(f.x) | ((unsigned)f2bf(f.y) << 16);
    o.y = (unsigned)f2bf(f.z) | ((unsigned)f2bf(f.w) << 16);
    *(uint2*)(dst + i) = o;
  }
}

// ---------------------------------------------------------------------------
// WMMA GEMM: C[z] = A[z] (MxK, bf16) x B[z]^T (NxK, bf16) [+ C]
// Wave tile: 32(M) x 64(N) -> 8 WMMAs / 32-wide K-step (2 A-frags x 4 B-frags)
// Block = 8 waves -> 256(M) x 64(N). Latency hidden by the compiler's
// load/WMMA software pipelining + 8-wave occupancy (prefetch variants only
// added overhead per the instruction-level evidence).
// ---------------------------------------------------------------------------
template<bool C_BF16, bool ADD>
__global__ __launch_bounds__(256) void gemm_bf16_wmma(
    const unsigned short* __restrict__ A,
    const unsigned short* __restrict__ Bp,
    void* __restrict__ Cp,
    int K, int lda, int ldb, int ldc,
    long strideA, long strideB,
    long strideC0, long strideC1, int czDiv)
{
  const int z    = blockIdx.z;
  const int wave = threadIdx.x >> 5;
  const int lane = threadIdx.x & 31;
  const int half = lane >> 4;           // K-half select (ISA A-frag layout)
  const int l16  = lane & 15;
  const int mbase = (blockIdx.y * 8 + wave) << 5;   // 32 rows per wave
  const int nbase = blockIdx.x << 6;                // 64 cols per block

  const unsigned short* Ab = A  + (long)z * strideA;
  const unsigned short* Bb = Bp + (long)z * strideB;
  const long coff = (long)(z / czDiv) * strideC0 + (long)(z % czDiv) * strideC1;

  f32x8 acc[2][4] = {};

  const unsigned short* arow0 = Ab + (long)(mbase + l16) * lda;
  const unsigned short* arow1 = Ab + (long)(mbase + 16 + l16) * lda;
  const unsigned short* brow[4];
#pragma unroll
  for (int t = 0; t < 4; ++t)
    brow[t] = Bb + (long)(nbase + t * 16 + l16) * ldb;

  for (int k0 = 0; k0 < K; k0 += 32) {
    Frag a0, a1;
    a0.q[0] = *(const uint4*)(arow0 + k0 + half * 8);
    a0.q[1] = *(const uint4*)(arow0 + k0 + 16 + half * 8);
    a1.q[0] = *(const uint4*)(arow1 + k0 + half * 8);
    a1.q[1] = *(const uint4*)(arow1 + k0 + 16 + half * 8);
#pragma unroll
    for (int t = 0; t < 4; ++t) {
      Frag b;
      b.q[0] = *(const uint4*)(brow[t] + k0 + half * 8);
      b.q[1] = *(const uint4*)(brow[t] + k0 + 16 + half * 8);
      acc[0][t] = __builtin_amdgcn_wmma_f32_16x16x32_bf16(
          false, a0.v, false, b.v, (short)0, acc[0][t], false, false);
      acc[1][t] = __builtin_amdgcn_wmma_f32_16x16x32_bf16(
          false, a1.v, false, b.v, (short)0, acc[1][t], false, false);
    }
  }

#pragma unroll
  for (int mi = 0; mi < 2; ++mi) {
#pragma unroll
    for (int t = 0; t < 4; ++t) {
#pragma unroll
      for (int r = 0; r < 8; ++r) {
        const int row = mbase + mi * 16 + r + half * 8;  // lanes16-31 -> M=r+8
        const int col = nbase + t * 16 + l16;
        const long ci = coff + (long)row * ldc + col;
        float v = acc[mi][t][r];
        if (C_BF16) {
          unsigned short* C = (unsigned short*)Cp;
          if (ADD) v += bf2f(C[ci]);
          C[ci] = f2bf(v);
        } else {
          float* C = (float*)Cp;
          if (ADD) v += C[ci];
          C[ci] = v;
        }
      }
    }
  }
}

// ---------------------------------------------------------------------------
// Embedding gather: h[t, :] = embed[ids[t], :]
// ---------------------------------------------------------------------------
__global__ __launch_bounds__(256) void embed_gather(
    const int* __restrict__ ids, const float* __restrict__ emb,
    float* __restrict__ h)
{
  const long t  = blockIdx.x;
  const long id = ids[t];
  for (int d = threadIdx.x; d < DD; d += 256)
    h[t * DD + d] = emb[id * DD + d];
}

// ---------------------------------------------------------------------------
// RMSNorm -> bf16: out = x * rsqrt(mean(x^2)+eps) * w
// ---------------------------------------------------------------------------
__global__ __launch_bounds__(256) void rmsnorm_bf16(
    const float* __restrict__ x, const float* __restrict__ w,
    unsigned short* __restrict__ out, int D)
{
  __shared__ float red[256];
  const long r = blockIdx.x;
  const int tid = threadIdx.x;
  const float* row = x + r * D;
  float ss = 0.f;
  for (int d = tid; d < D; d += 256) { float v = row[d]; ss += v * v; }
  red[tid] = ss; __syncthreads();
  for (int s = 128; s > 0; s >>= 1) {
    if (tid < s) red[tid] += red[tid + s];
    __syncthreads();
  }
  const float inv = rsqrtf(red[0] / (float)D + 1e-6f);
  for (int d = tid; d < D; d += 256)
    out[r * D + d] = f2bf(row[d] * inv * w[d]);
}

// ---------------------------------------------------------------------------
// RoPE on q,k + split qkv into head-major bf16 buffers.
//   qb,kb: [B,H,S,HD]   vt: [B,H,HD,S] (transposed for PV = A*B^T WMMA)
// ---------------------------------------------------------------------------
__global__ __launch_bounds__(HDD) void rope_split(
    const float* __restrict__ qkv, const int* __restrict__ pos,
    unsigned short* __restrict__ qb, unsigned short* __restrict__ kb,
    unsigned short* __restrict__ vt)
{
  const int t = blockIdx.x;          // token index b*S+s
  const int h = blockIdx.y;
  const int d = threadIdx.x;         // 0..127
  const int s = t & (SS - 1);
  const int b = t >> 10;             // SS = 1024
  const int dm = d & 63;
  const float p = (float)pos[s];
  const float freq = __powf(10000.0f, -((float)dm) * (1.0f / 64.0f));
  const float ang = p * freq;
  const float c = __cosf(ang), sn = __sinf(ang);

  const float* base = qkv + (long)t * (3 * DD);
  const float qv = base[h * HDD + d];
  const float kv = base[DD + h * HDD + d];
  const float vv = base[2 * DD + h * HDD + d];
  const float qrot = (d < 64) ? -base[h * HDD + d + 64]      : base[h * HDD + d - 64];
  const float krot = (d < 64) ? -base[DD + h * HDD + d + 64] : base[DD + h * HDD + d - 64];
  const float qo = qv * c + qrot * sn;
  const float ko = kv * c + krot * sn;

  const long zi = (long)b * HH + h;
  qb[(zi * SS + s) * HDD + d] = f2bf(qo);
  kb[(zi * SS + s) * HDD + d] = f2bf(ko);
  vt[(zi * HDD + d) * SS + s] = f2bf(vv);
}

// ---------------------------------------------------------------------------
// Causal softmax row-wise: probs(bf16) = softmax(scale*scores masked)
// one block per row (b,h,q)
// ---------------------------------------------------------------------------
__global__ __launch_bounds__(256) void softmax_causal(
    const float* __restrict__ scores, unsigned short* __restrict__ probs,
    float scale)
{
  __shared__ float red[256];
  const long r = blockIdx.x;
  const int q = (int)(r & (SS - 1));
  const int tid = threadIdx.x;
  const float* srow = scores + r * SS;
  unsigned short* prow = probs + r * SS;

  float mx = -3.0e38f;
  for (int k = tid; k < SS; k += 256) {
    float v = (k <= q) ? srow[k] * scale : -3.0e38f;
    mx = fmaxf(mx, v);
  }
  red[tid] = mx; __syncthreads();
  for (int s = 128; s > 0; s >>= 1) {
    if (tid < s) red[tid] = fmaxf(red[tid], red[tid + s]);
    __syncthreads();
  }
  mx = red[0]; __syncthreads();

  float sum = 0.f;
  for (int k = tid; k < SS; k += 256)
    sum += (k <= q) ? __expf(srow[k] * scale - mx) : 0.f;
  red[tid] = sum; __syncthreads();
  for (int s = 128; s > 0; s >>= 1) {
    if (tid < s) red[tid] += red[tid + s];
    __syncthreads();
  }
  const float inv = 1.f / red[0];

  for (int k = tid; k < SS; k += 256) {
    float v = (k <= q) ? __expf(srow[k] * scale - mx) * inv : 0.f;
    prow[k] = f2bf(v);
  }
}

// ---------------------------------------------------------------------------
// SwiGLU: act = silu(gate) * up  -> bf16
// ---------------------------------------------------------------------------
__global__ __launch_bounds__(256) void swiglu_bf16(
    const float* __restrict__ gu, unsigned short* __restrict__ act)
{
  const long idx = (long)blockIdx.x * 256 + threadIdx.x;   // over TT*II
  const long m = idx / II;
  const long i = idx - m * II;
  const float g = gu[m * (2 * II) + i];
  const float u = gu[m * (2 * II) + II + i];
  const float sg = g / (1.f + __expf(-g));
  act[idx] = f2bf(sg * u);
}

// ---------------------------------------------------------------------------
extern "C" void kernel_launch(void* const* d_in, const int* in_sizes, int n_in,
                              void* d_out, int out_size, void* d_ws, size_t ws_size,
                              hipStream_t stream) {
  (void)in_sizes; (void)n_in; (void)out_size; (void)ws_size;
  const int*   ids   = (const int*)  d_in[0];
  const int*   pos   = (const int*)  d_in[1];
  const float* emb   = (const float*)d_in[2];
  const float* Wqkv  = (const float*)d_in[3];
  const float* Wo    = (const float*)d_in[4];
  const float* Wgu   = (const float*)d_in[5];
  const float* Wd    = (const float*)d_in[6];
  const float* ln1   = (const float*)d_in[7];
  const float* ln2   = (const float*)d_in[8];
  const float* normw = (const float*)d_in[9];
  const float* lmh   = (const float*)d_in[10];
  float* out = (float*)d_out;

  char* ws = (char*)d_ws;
  size_t off = 0;
  auto carve = [&](size_t bytes) -> char* {
    char* p = ws + off;
    off += (bytes + 255) & ~(size_t)255;
    return p;
  };
  float*          h      = (float*)         carve((size_t)TT * DD * 4);        // 16 MB
  unsigned short* xbf    = (unsigned short*)carve((size_t)TT * DD * 2);        //  8 MB
  float*          qkv    = (float*)         carve((size_t)TT * 3 * DD * 4);    // 48 MB
  unsigned short* qb     = (unsigned short*)carve((size_t)BB * HH * SS * HDD * 2);
  unsigned short* kb     = (unsigned short*)carve((size_t)BB * HH * SS * HDD * 2);
  unsigned short* vt     = (unsigned short*)carve((size_t)BB * HH * HDD * SS * 2);
  unsigned short* attnbf = (unsigned short*)carve((size_t)TT * DD * 2);        //  8 MB
  unsigned short* wbf    = (unsigned short*)carve((size_t)VV * DD * 2);        // 131 MB weight staging
  float*          scores = (float*)         carve((size_t)BB * HH * SS * SS * 4); // 128 MB (reused as gu)
  unsigned short* probs  = (unsigned short*)carve((size_t)BB * HH * SS * SS * 2); //  64 MB (reused as act)
  float*          gu     = scores;          // 2048x11264 fp32 = 92 MB <= 128 MB
  unsigned short* act    = probs;           // 2048x5632 bf16 = 23 MB <= 64 MB

  auto cvtW = [&](const float* w, long n) {
    cvt_f32_bf16<<<(int)((n + 1023) / 1024), 256, 0, stream>>>(w, wbf, n);
  };

  embed_gather<<<TT, 256, 0, stream>>>(ids, emb, h);

  for (int l = 0; l < LL; ++l) {
    // x = rmsnorm(h, ln1) in bf16
    rmsnorm_bf16<<<TT, 256, 0, stream>>>(h, ln1 + (size_t)l * DD, xbf, DD);
    // qkv = x @ Wqkv^T   (2048 x 6144 x 2048)
    cvtW(Wqkv + (size_t)l * 3 * DD * DD, (long)3 * DD * DD);
    gemm_bf16_wmma<false, false><<<dim3(3 * DD / 64, TT / 256, 1), 256, 0, stream>>>(
        xbf, wbf, qkv, DD, DD, DD, 3 * DD, 0, 0, 0, 0, 1);
    // rope + head-major split
    rope_split<<<dim3(TT, HH), HDD, 0, stream>>>(qkv, pos, qb, kb, vt);
    // scores[z] = q @ k^T  per (b,h)   (1024 x 1024 x 128, z=32)
    gemm_bf16_wmma<false, false><<<dim3(SS / 64, SS / 256, BB * HH), 256, 0, stream>>>(
        qb, kb, scores, HDD, HDD, HDD, SS,
        (long)SS * HDD, (long)SS * HDD, (long)SS * SS, 0, 1);
    softmax_causal<<<BB * HH * SS, 256, 0, stream>>>(scores, probs, 0.08838834764831845f);
    // attn[z] = probs @ vT^T -> attnbf[B,S,H*HD] bf16  (1024 x 128 x 1024)
    gemm_bf16_wmma<true, false><<<dim3(HDD / 64, SS / 256, BB * HH), 256, 0, stream>>>(
        probs, vt, attnbf, SS, SS, SS, HH * HDD,
        (long)SS * SS, (long)HDD * SS, (long)SS * HH * HDD, (long)HDD, HH);
    // h += attn @ Wo^T
    cvtW(Wo + (size_t)l * DD * DD, (long)DD * DD);
    gemm_bf16_wmma<false, true><<<dim3(DD / 64, TT / 256, 1), 256, 0, stream>>>(
        attnbf, wbf, h, DD, DD, DD, DD, 0, 0, 0, 0, 1);
    // MLP
    rmsnorm_bf16<<<TT, 256, 0, stream>>>(h, ln2 + (size_t)l * DD, xbf, DD);
    cvtW(Wgu + (size_t)l * 2 * II * DD, (long)2 * II * DD);
    gemm_bf16_wmma<false, false><<<dim3(2 * II / 64, TT / 256, 1), 256, 0, stream>>>(
        xbf, wbf, gu, DD, DD, DD, 2 * II, 0, 0, 0, 0, 1);
    swiglu_bf16<<<(int)(((size_t)TT * II) / 256), 256, 0, stream>>>(gu, act);
    cvtW(Wd + (size_t)l * DD * II, (long)DD * II);
    gemm_bf16_wmma<false, true><<<dim3(DD / 64, TT / 256, 1), 256, 0, stream>>>(
        act, wbf, h, II, II, II, DD, 0, 0, 0, 0, 1);
  }

  // final norm + lm_head (2048 x 32000 x 2048)
  rmsnorm_bf16<<<TT, 256, 0, stream>>>(h, normw, xbf, DD);
  cvtW(lmh, (long)VV * DD);
  gemm_bf16_wmma<false, false><<<dim3(VV / 64, TT / 256, 1), 256, 0, stream>>>(
      xbf, wbf, out, DD, DD, DD, VV, 0, 0, 0, 0, 1);
}
